// CrossAttention_Sp_88459146429000
// MI455X (gfx1250) — compile-verified, matrix operand order checked by reading, not run on hardware
//
#include <hip/hip_runtime.h>

// MI455X (gfx1250). All GEMMs on v_wmma_f32_16x16x32_bf16 (wave32, f32 acc).
// Compute bound (~135 GFLOP vs ~200MB HBM => >10x compute bound @ 23.3 TB/s).
// Fragments are two b128 loads (operands pre-swizzled into K-panels);
// k_gemm ping-pongs LDS tiles so global loads overlap WMMA; B-fragment loads
// are batched ahead of the WMMA burst so DS waits overlap matrix issue.

typedef __attribute__((ext_vector_type(16))) __bf16 v16bf;
typedef __attribute__((ext_vector_type(8)))  float  v8f;

#define DIMX 768
#define NPTS 1024
#define NBAT 16
#define ROWS (NBAT*NPTS)          // 16384

// Lane's fragment = two contiguous 16B runs of a 32-element K row:
//   lo at +hi8, hi at +16+hi8, hi8 = 8*(lane>=16)   (ISA 7.12.2 16-bit layout)
__device__ __forceinline__ v16bf frag_ld(const __bf16* rowbase) {
  int lane = threadIdx.x & 31;
  const __bf16* p = rowbase + ((lane & 16) ? 8 : 0);
  union { struct { uint4 lo, hi; } u; v16bf f; } q;
  q.u.lo = *(const uint4*)(p);
  q.u.hi = *(const uint4*)(p + 16);
  return q.f;
}

__device__ __forceinline__ v8f wmma_bf16(v16bf a, v16bf b, v8f c) {
  return __builtin_amdgcn_wmma_f32_16x16x32_bf16(false, a, false, b,
                                                 (short)0, c, false, false);
}

__device__ __forceinline__ v8f vzero8() {
  v8f z;
#pragma unroll
  for (int i = 0; i < 8; ++i) z[i] = 0.f;
  return z;
}

// ---------------- conversion kernels ----------------
// B operands use K-panel layout: Bp[((k>>5)*N + n)*32 + (k&31)] = B[k][n]

__global__ __launch_bounds__(256) void k_cvt_xy(
    const float* __restrict__ x, const float* __restrict__ y,
    __bf16* __restrict__ xb, __bf16* __restrict__ yTp, __bf16* __restrict__ ybp) {
  size_t i = (size_t)blockIdx.x * 256 + threadIdx.x;
  if (i >= (size_t)ROWS * DIMX) return;
  xb[i] = (__bf16)x[i];                       // A operand: row-major
  float yv = y[i];
  size_t row = i / DIMX;
  int d = (int)(i % DIMX);
  int bb = (int)(row >> 10), m = (int)(row & 1023);
  size_t bs = (size_t)bb * NPTS * DIMX;       // panel batch stride == K*N
  yTp[bs + ((size_t)((d >> 5) * NPTS + m)) * 32 + (d & 31)] = (__bf16)yv;  // B[k=d][n=m]
  ybp[bs + ((size_t)((m >> 5) * DIMX + d)) * 32 + (m & 31)] = (__bf16)yv;  // B[k=m][n=d]
}

__global__ __launch_bounds__(256) void k_cvt_U(
    const float* __restrict__ U, __bf16* __restrict__ Ubp, __bf16* __restrict__ Utbp) {
  int i = blockIdx.x * 256 + threadIdx.x;
  if (i >= DIMX * DIMX) return;
  int r = i / DIMX, c = i % DIMX;
  float v = U[i];
  Ubp[((size_t)((r >> 5) * DIMX + c)) * 32 + (r & 31)] = (__bf16)v;   // (.)@U
  Utbp[((size_t)((c >> 5) * DIMX + r)) * 32 + (c & 31)] = (__bf16)v;  // x@U^T
}

// ---------------- pos_score = softmax(coords @ pos_emb) ----------------

__device__ __forceinline__ float blk_red_max(float v, float* sb) {
#pragma unroll
  for (int o = 16; o > 0; o >>= 1) v = fmaxf(v, __shfl_xor(v, o, 32));
  if ((threadIdx.x & 31) == 0) sb[threadIdx.x >> 5] = v;
  __syncthreads();
  float r = sb[0];
  for (int i = 1; i < 8; ++i) r = fmaxf(r, sb[i]);
  __syncthreads();
  return r;
}

__device__ __forceinline__ float blk_red_sum(float v, float* sb) {
#pragma unroll
  for (int o = 16; o > 0; o >>= 1) v += __shfl_xor(v, o, 32);
  if ((threadIdx.x & 31) == 0) sb[threadIdx.x >> 5] = v;
  __syncthreads();
  float r = 0.f;
  for (int i = 0; i < 8; ++i) r += sb[i];
  __syncthreads();
  return r;
}

__global__ __launch_bounds__(256) void k_pos(
    const float* __restrict__ coords, const float* __restrict__ pos_emb,
    float* __restrict__ pos_score) {
  __shared__ float pe[6];
  __shared__ float sb[8];
  int n = blockIdx.x;
  if (threadIdx.x < 6) pe[threadIdx.x] = pos_emb[n * 6 + threadIdx.x];
  __syncthreads();
  float vals[4];
  float mx = -1e30f;
#pragma unroll
  for (int j = 0; j < 4; ++j) {
    int m = threadIdx.x + j * 256;
    const float* c = coords + ((size_t)n * NPTS + m) * 6;
    float s = 0.f;
#pragma unroll
    for (int q = 0; q < 6; ++q) s += c[q] * pe[q];
    vals[j] = s;
    mx = fmaxf(mx, s);
  }
  mx = blk_red_max(mx, sb);
  float sum = 0.f;
#pragma unroll
  for (int j = 0; j < 4; ++j) { vals[j] = __expf(vals[j] - mx); sum += vals[j]; }
  sum = blk_red_sum(sum, sb);
  float inv = 1.f / sum;
#pragma unroll
  for (int j = 0; j < 4; ++j)
    pos_score[(size_t)n * NPTS + threadIdx.x + j * 256] = vals[j] * inv;
}

// ------- generic bf16 WMMA GEMM, double-buffered: C = A[MxK] * Bpanel[KxN] ---
// mode 0: out1 = bf16(acc*S1[n]^2), out2 = bf16(acc*S2[n]^2);  mode 1: out1 = bf16(acc)

__global__ __launch_bounds__(256) void k_gemm(
    const __bf16* __restrict__ A, const __bf16* __restrict__ Bp,
    int M, int N, int K,
    const float* __restrict__ S1, const float* __restrict__ S2,
    __bf16* __restrict__ out1, __bf16* __restrict__ out2, int mode) {
  __shared__ __bf16 As[2][128 * 32];     // [row][k]
  __shared__ __bf16 Bs[2][128 * 32];     // [n][k] (fragment-ready)
  int nb = N >> 7;
  int bx = blockIdx.x % nb, by = blockIdx.x / nb;
  int tid = threadIdx.x, wave = tid >> 5, lane = tid & 31;
  int row0 = by << 7, col0 = bx << 7;
  v8f acc[8];
#pragma unroll
  for (int i = 0; i < 8; ++i) acc[i] = vzero8();

  int hr = tid >> 1, hc = (tid & 1) << 4;      // 2 threads per 32-elem row
  const __bf16* Ag = A + (size_t)(row0 + hr) * K + hc;
  const __bf16* Bg = Bp + (size_t)(col0 + hr) * 32 + hc;
  uint4 ra0, ra1, rb0, rb1;
  auto ld = [&](int k0) {
    const uint4* sa = (const uint4*)(Ag + k0);
    ra0 = sa[0]; ra1 = sa[1];
    const uint4* sb = (const uint4*)(Bg + (size_t)(k0 >> 5) * N * 32);
    rb0 = sb[0]; rb1 = sb[1];
  };
  auto st = [&](int buf) {
    uint4* da = (uint4*)(As[buf] + hr * 32 + hc);
    da[0] = ra0; da[1] = ra1;
    uint4* db = (uint4*)(Bs[buf] + hr * 32 + hc);
    db[0] = rb0; db[1] = rb1;
  };

  int nk = K >> 5;
  ld(0); st(0);
  __syncthreads();
  for (int ks = 0; ks < nk; ++ks) {
    int cur = ks & 1;
    if (ks + 1 < nk) ld((ks + 1) << 5);          // global -> regs, overlaps WMMA
    v16bf af = frag_ld(As[cur] + (wave * 16 + (lane & 15)) * 32);
    v16bf bfr[8];
#pragma unroll
    for (int ct = 0; ct < 8; ++ct)
      bfr[ct] = frag_ld(Bs[cur] + ((ct << 4) + (lane & 15)) * 32);
#pragma unroll
    for (int ct = 0; ct < 8; ++ct)
      acc[ct] = wmma_bf16(af, bfr[ct], acc[ct]);
    if (ks + 1 < nk) st(cur ^ 1);
    __syncthreads();
  }

  int rb = (wave << 4) + ((lane >> 4) << 3);     // C/D: M = i + 8*(lane>=16)
#pragma unroll
  for (int ct = 0; ct < 8; ++ct) {
    int col = col0 + (ct << 4) + (lane & 15);
    float sa = 1.f, sb2 = 1.f;
    if (mode == 0) { float a = S1[col], b2 = S2[col]; sa = a * a; sb2 = b2 * b2; }
#pragma unroll
    for (int i = 0; i < 8; ++i) {
      size_t idx = (size_t)(row0 + rb + i) * N + col;
      float v = acc[ct][i];
      if (mode == 0) { out1[idx] = (__bf16)(v * sa); out2[idx] = (__bf16)(v * sb2); }
      else           { out1[idx] = (__bf16)v; }
    }
  }
}

// ------- fused attention: scores -> softmax/entropy/route -> attn@y -------
// 512 threads (16 waves): each wave owns 64 m-cols in phase B (both heads
// per B-fragment load) and 48 d-cols in phase E.

#define SMEM_XS   0                         // 2 x 16 x 768 bf16 = 49152 B
#define SMEM_S    49152                     // 2 x 16 x 1024 f32 = 131072 B
#define SMEM_RED  (49152 + 131072)          // 128 f32
#define SMEM_BYTES (49152 + 131072 + 512)

__global__ __launch_bounds__(512) void k_attn(
    const __bf16* __restrict__ X1, const __bf16* __restrict__ X2,
    const __bf16* __restrict__ ybp, const __bf16* __restrict__ yTp,
    const float* __restrict__ pos_score,
    const float* __restrict__ gating, const float* __restrict__ temp,
    float* __restrict__ out, float* __restrict__ heat) {
  extern __shared__ char smem[];
  __bf16* Xs = (__bf16*)(smem + SMEM_XS);      // phase A/B: query tiles
  float*  S  = (float*)(smem + SMEM_S);        // [2][16][1024] logits
  float*  red = (float*)(smem + SMEM_RED);
  __bf16* AC = (__bf16*)(smem + SMEM_XS);      // phase D/E: attn_c (reuses Xs)
  float* red_max = red, *red_sum = red + 32, *red_ent = red + 64, *red_ch = red + 96;

  int b = blockIdx.x >> 6, tile = blockIdx.x & 63;
  int n0 = tile << 4;
  int tid = threadIdx.x, w = tid >> 5, lane = tid & 31;

  // Phase A: stage 16-row query tiles (both heads) via b128 copies
  {
    const uint4* s1 = (const uint4*)(X1 + ((size_t)(b * NPTS + n0)) * DIMX);
    const uint4* s2 = (const uint4*)(X2 + ((size_t)(b * NPTS + n0)) * DIMX);
    uint4* d = (uint4*)Xs;
    for (int i = tid; i < (16 * DIMX) / 8; i += 512) {
      d[i] = s1[i];
      d[(16 * DIMX) / 8 + i] = s2[i];
    }
  }
  __syncthreads();

  // Phase B: S[h][r][m] = scale*(x_h @ y^T); wave w owns m in [w*64, w*64+64)
  const float scl = 0.0360843918243516f;  // 768^-0.5
  const __bf16* YTb = yTp + (size_t)b * DIMX * NPTS;
  {
    int mq = (w << 6) + (lane & 15);
    v8f acc0[4], acc1[4];
#pragma unroll
    for (int i = 0; i < 4; ++i) { acc0[i] = vzero8(); acc1[i] = vzero8(); }
    for (int ks = 0; ks < DIMX / 32; ++ks) {
      if (ks + 1 < DIMX / 32)
        __builtin_prefetch(YTb + ((size_t)(ks + 1) * NPTS + mq) * 32, 0, 3);
      v16bf af0 = frag_ld(Xs + (size_t)(lane & 15) * DIMX + ks * 32);
      v16bf af1 = frag_ld(Xs + 16 * DIMX + (size_t)(lane & 15) * DIMX + ks * 32);
      v16bf bfr[4];
#pragma unroll
      for (int ct = 0; ct < 4; ++ct)
        bfr[ct] = frag_ld(YTb + ((size_t)ks * NPTS + mq + (ct << 4)) * 32);
#pragma unroll
      for (int ct = 0; ct < 4; ++ct) {
        acc0[ct] = wmma_bf16(af0, bfr[ct], acc0[ct]);
        acc1[ct] = wmma_bf16(af1, bfr[ct], acc1[ct]);
      }
    }
    int rb = (lane >> 4) << 3;
#pragma unroll
    for (int ct = 0; ct < 4; ++ct) {
      int m = (w << 6) + (ct << 4) + (lane & 15);
#pragma unroll
      for (int i = 0; i < 8; ++i) {
        S[(rb + i) * NPTS + m] = acc0[ct][i] * scl;
        S[16 * NPTS + (rb + i) * NPTS + m] = acc1[ct][i] * scl;
      }
    }
  }
  __syncthreads();

  float g = 1.f / (1.f + __expf(-gating[0]));
  float tq = temp[0];

  // Phase C: per (head,row): max, sumexp, entropy; 16 lanes per row
  {
    int p = tid >> 4, h = p & 1, r = p >> 1, sub = tid & 15;
    float* Sr = S + (h * 16 + r) * NPTS;
    const float* pr = pos_score + (size_t)(n0 + r) * NPTS;
    float mx = -1e30f;
    for (int m = sub; m < NPTS; m += 16) mx = fmaxf(mx, Sr[m]);
#pragma unroll
    for (int o = 1; o < 16; o <<= 1) mx = fmaxf(mx, __shfl_xor(mx, o, 32));
    float sum = 0.f;
    for (int m = sub; m < NPTS; m += 16) sum += __expf(Sr[m] - mx);
#pragma unroll
    for (int o = 1; o < 16; o <<= 1) sum += __shfl_xor(sum, o, 32);
    float inv = 1.f / sum;
    float ent = 0.f;
    for (int m = sub; m < NPTS; m += 16) {
      float a = (1.f - g) * __expf(Sr[m] - mx) * inv + g * pr[m];
      ent -= a * __logf(a + 1e-8f);
    }
#pragma unroll
    for (int o = 1; o < 16; o <<= 1) ent += __shfl_xor(ent, o, 32);
    if (sub == 0) { red_max[h * 16 + r] = mx; red_sum[h * 16 + r] = inv; red_ent[h * 16 + r] = ent; }
  }
  __syncthreads();
  if (tid < 16) {     // route + heat_map
    float h0 = 2.f - 2.f / (1.f + __expf(-tq * red_ent[tid]));
    float h1 = 2.f - 2.f / (1.f + __expf(-tq * red_ent[16 + tid]));
    bool fg = h0 >= h1;
    heat[(size_t)b * NPTS + n0 + tid] = fg ? h0 : h1;
    red_ch[tid] = fg ? 0.f : 1.f;
  }
  __syncthreads();

  // Phase D: routed attention row -> bf16 AC (packed 8-wide stores)
  {
    int r = tid >> 5, seg = tid & 31;
    int hsel = (int)red_ch[r];
    float mx = red_max[hsel * 16 + r], inv = red_sum[hsel * 16 + r];
    const float* Sr = S + (hsel * 16 + r) * NPTS;
    const float* pr = pos_score + (size_t)(n0 + r) * NPTS;
    for (int m0 = seg * 32; m0 < seg * 32 + 32; m0 += 8) {
      union { __bf16 h[8]; uint4 q; } pk;
#pragma unroll
      for (int j = 0; j < 8; ++j) {
        float a = (1.f - g) * __expf(Sr[m0 + j] - mx) * inv + g * pr[m0 + j];
        pk.h[j] = (__bf16)a;
      }
      *(uint4*)(AC + r * NPTS + m0) = pk.q;
    }
  }
  __syncthreads();

  // Phase E: out tile = AC[16x1024] @ y_b[1024x768]; wave w owns 48 d-cols
  {
    v8f acc[3];
#pragma unroll
    for (int i = 0; i < 3; ++i) acc[i] = vzero8();
    const __bf16* Ybb = ybp + (size_t)b * NPTS * DIMX;
    int cq = w * 48 + (lane & 15);
    for (int ks = 0; ks < NPTS / 32; ++ks) {
      if (ks + 1 < NPTS / 32)
        __builtin_prefetch(Ybb + ((size_t)(ks + 1) * DIMX + cq) * 32, 0, 3);
      v16bf af = frag_ld(AC + (size_t)(lane & 15) * NPTS + ks * 32);
      v16bf bfr[3];
#pragma unroll
      for (int ct = 0; ct < 3; ++ct)
        bfr[ct] = frag_ld(Ybb + ((size_t)ks * DIMX + cq + (ct << 4)) * 32);
#pragma unroll
      for (int ct = 0; ct < 3; ++ct)
        acc[ct] = wmma_bf16(af, bfr[ct], acc[ct]);
    }
    int rb = (lane >> 4) << 3;
#pragma unroll
    for (int ct = 0; ct < 3; ++ct) {
      int col = w * 48 + (ct << 4) + (lane & 15);
#pragma unroll
      for (int i = 0; i < 8; ++i)
        out[((size_t)(b * NPTS + n0 + rb + i)) * DIMX + col] = acc[ct][i];
    }
  }
}

// ---------------- host launch ----------------

extern "C" void kernel_launch(void* const* d_in, const int* in_sizes, int n_in,
                              void* d_out, int out_size, void* d_ws, size_t ws_size,
                              hipStream_t stream) {
  const float* x       = (const float*)d_in[0];
  const float* y       = (const float*)d_in[1];
  const float* coords  = (const float*)d_in[2];
  const float* U       = (const float*)d_in[3];
  const float* S1      = (const float*)d_in[4];
  const float* S2      = (const float*)d_in[5];
  const float* gating  = (const float*)d_in[6];
  const float* temp    = (const float*)d_in[7];
  const float* pos_emb = (const float*)d_in[8];

  float* out  = (float*)d_out;
  float* heat = out + (size_t)NBAT * NPTS * DIMX;

  const size_t EL = (size_t)ROWS * DIMX;          // 12,582,912 elements
  char* ws = (char*)d_ws;
  size_t off = 0;
  auto take = [&](size_t bytes) { void* p = ws + off; off += (bytes + 255) & ~(size_t)255; return p; };

  __bf16* xb   = (__bf16*)take(EL * 2);           // x bf16 (reused for X1 later)
  __bf16* yTp  = (__bf16*)take(EL * 2);           // y^T panels [b][kp*1024+m][32]
  __bf16* ybp  = (__bf16*)take(EL * 2);           // y panels   [b][kp*768+d][32]
  __bf16* Ubp  = (__bf16*)take((size_t)DIMX * DIMX * 2);
  __bf16* Utbp = (__bf16*)take((size_t)DIMX * DIMX * 2);
  __bf16* xs1  = (__bf16*)take(EL * 2);           // xu * S1^2 (row-major)
  __bf16* xs2  = (__bf16*)take(EL * 2);           // xu * S2^2
  __bf16* X2b  = (__bf16*)take(EL * 2);           // x2 queries
  float*  posS = (float*)take((size_t)NPTS * NPTS * 4);
  __bf16* X1b  = xb;                              // reuse: xb dead after gemm1

  (void)in_sizes; (void)n_in; (void)out_size; (void)ws_size;

  // conversions (build A row-major + B panel layouts)
  k_cvt_xy<<<(int)((EL + 255) / 256), 256, 0, stream>>>(x, y, xb, yTp, ybp);
  k_cvt_U<<<(DIMX * DIMX + 255) / 256, 256, 0, stream>>>(U, Ubp, Utbp);

  // pos softmax
  k_pos<<<NPTS, 256, 0, stream>>>(coords, pos_emb, posS);

  // GEMM1: xu = x @ U^T, fused S^2 scaling -> xs1, xs2 (bf16)
  k_gemm<<<(ROWS / 128) * (DIMX / 128), 256, 0, stream>>>(
      xb, Utbp, ROWS, DIMX, DIMX, S1, S2, xs1, xs2, 0);
  // GEMM2: x1 = xs1 @ U ; x2 = xs2 @ U
  k_gemm<<<(ROWS / 128) * (DIMX / 128), 256, 0, stream>>>(
      xs1, Ubp, ROWS, DIMX, DIMX, nullptr, nullptr, X1b, nullptr, 1);
  k_gemm<<<(ROWS / 128) * (DIMX / 128), 256, 0, stream>>>(
      xs2, Ubp, ROWS, DIMX, DIMX, nullptr, nullptr, X2b, nullptr, 1);

  // fused attention (needs >64KB dynamic LDS; WGP has 320KB)
  (void)hipFuncSetAttribute((const void*)k_attn,
                            hipFuncAttributeMaxDynamicSharedMemorySize,
                            (int)SMEM_BYTES);
  k_attn<<<NBAT * (NPTS / 16), 512, SMEM_BYTES, stream>>>(
      X1b, X2b, ybp, yTp, posS, gating, temp, out, heat);
}
